// MultiHeadLatentAttention_66958540145209
// MI455X (gfx1250) — matrix operand order repaired and checked by reading, not compiled, not run on hardware
//
#include <hip/hip_runtime.h>
#include <cstdint>

// ---------------------------------------------------------------------------
// MLA forward for MI455X (gfx1250): bf16 WMMA everywhere, f32 accumulate.
// GEMM tile staging: double-buffered CDNA5 async global->LDS copies
// (GLOBAL_LOAD_ASYNC_TO_LDS_B128 / ASYNCcnt), copy latency hidden behind WMMA.
// ---------------------------------------------------------------------------

typedef __bf16 bf16;
typedef __attribute__((ext_vector_type(16))) __bf16 v16bf;
typedef __attribute__((ext_vector_type(8)))  float  v8f;

union FragAB { v16bf v; uint4 u[2]; };   // 16 bf16 = 32B
union FragC  { v8f   v; float f[8]; };   // 8 f32

// --- CDNA5 async copy helpers ---------------------------------------------
// Flat->LDS mapping on gfx1250 truncates to addr[31:0], so the low 32 bits of
// a generic pointer to __shared__ memory are the LDS byte address.
__device__ __forceinline__ unsigned lds_off(const void* p) {
  return (unsigned)(unsigned long long)p;
}
// GLOBAL_LOAD_ASYNC_TO_LDS_B128: per-lane 16B copy, tracked by ASYNCcnt.
__device__ __forceinline__ void async_cp16(unsigned lds_addr, const void* gaddr) {
  asm volatile("global_load_async_to_lds_b128 %0, %1, off"
               :: "v"(lds_addr), "v"(gaddr) : "memory");
}
// Per-wave async ops complete in order: waiting ASYNCcnt<=4 retires the
// previous tile's 4 copies while the next tile's 4 stay in flight.
__device__ __forceinline__ void wait_async_le4() {
  asm volatile("s_wait_asynccnt 0x4" ::: "memory");
}
__device__ __forceinline__ void wait_async_le0() {
  asm volatile("s_wait_asynccnt 0x0" ::: "memory");
}

// ======================== generic bf16 WMMA GEMM ===========================
// C[m,n] = sum_k A[m,k] * B[n,k] + bias[n]   (A: MxK, B: NxK, both row-major)
// Block: 256 threads = 8 waves; tile BM=128, BN=128, BK=32.
// Wave (wm = w&3, wn = w>>2) owns a 32x64 sub-tile = 2x4 grid of 16x16 WMMAs.
// gridDim.z batches heads via element strides sA/sB/sC.
#define BM 128
#define BN 128
#define BK 32
#define LDSA 40   // 32 data + 8 pad bf16 (80B = 20 banks, 16B aligned rows)

template <typename OutT>
__global__ __launch_bounds__(256) void gemm_bf16_wmma(
    const bf16* __restrict__ A, long lda, long sA,
    const bf16* __restrict__ Bm, long ldb, long sB,
    const float* __restrict__ bias,
    OutT* __restrict__ C, long ldc, long sC,
    int M, int N, int K)
{
  __shared__ bf16 As[2][BM * LDSA];
  __shared__ bf16 Bs[2][BN * LDSA];
  A  += (long)blockIdx.z * sA;
  Bm += (long)blockIdx.z * sB;
  C  += (long)blockIdx.z * sC;
  const int m0 = blockIdx.y * BM, n0 = blockIdx.x * BN;
  const int tid = threadIdx.x, lane = tid & 31, wave = tid >> 5;
  const int wm = wave & 3, wn = wave >> 2;
  const int khalf = lane >> 4, l16 = lane & 15;
  const int srow = tid >> 1, shalf = (tid & 1) * 16;   // staging: half-row per thread
  // OOB rows are clamped: their garbage lands only in LDS rows feeding
  // output rows/cols that the epilogue never stores.
  const int arow = (m0 + srow < M) ? (m0 + srow) : (M - 1);
  const int brow = (n0 + srow < N) ? (n0 + srow) : (N - 1);
  const unsigned ldsA0 = lds_off(&As[0][srow * LDSA + shalf]);
  const unsigned ldsB0 = lds_off(&Bs[0][srow * LDSA + shalf]);
  const unsigned bufBytes = (unsigned)(BM * LDSA * sizeof(bf16));
  const bf16* gArow = A + (long)arow * lda + shalf;
  const bf16* gBrow = Bm + (long)brow * ldb + shalf;

  FragC acc[2][4];
#pragma unroll
  for (int i = 0; i < 2; i++)
#pragma unroll
    for (int j = 0; j < 4; j++)
#pragma unroll
      for (int v = 0; v < 8; v++) acc[i][j].f[v] = 0.f;

  // prologue: stage K-tile 0 into buffer 0 (4 x 16B per thread, async)
  async_cp16(ldsA0,      gArow);
  async_cp16(ldsA0 + 16, gArow + 8);
  async_cp16(ldsB0,      gBrow);
  async_cp16(ldsB0 + 16, gBrow + 8);

  const int nk = K / BK;
  for (int t = 0; t < nk; t++) {
    const int buf = t & 1;
    if (t + 1 < nk) {
      // issue next tile into the other buffer, then retire current tile's
      // copies (ASYNCcnt <= 4): latency hidden behind this tile's WMMAs
      const unsigned off = (buf ^ 1) * bufBytes;
      const bf16* ga = gArow + (long)(t + 1) * BK;
      const bf16* gb = gBrow + (long)(t + 1) * BK;
      async_cp16(ldsA0 + off,      ga);
      async_cp16(ldsA0 + off + 16, ga + 8);
      async_cp16(ldsB0 + off,      gb);
      async_cp16(ldsB0 + off + 16, gb + 8);
      wait_async_le4();
    } else {
      wait_async_le0();
    }
    __syncthreads();                 // current tile visible to all waves

    const bf16* Asb = As[buf];
    const bf16* Bsb = Bs[buf];
    // A frag (16x32 bf16): lanes 0-15 = rows, khalf selects K {0..7,16..23} vs {8..15,24..31}
    FragAB af[2], bfr[4];
#pragma unroll
    for (int i = 0; i < 2; i++) {
      const bf16* p = &Asb[(wm * 32 + i * 16 + l16) * LDSA + khalf * 8];
      af[i].u[0] = *(const uint4*)p;
      af[i].u[1] = *(const uint4*)(p + 16);
    }
    // B frag (32x16): lane holds column n = l16, khalf selects K {0..7,16..23} vs {8..15,24..31}
#pragma unroll
    for (int j = 0; j < 4; j++) {
      const bf16* p = &Bsb[(wn * 64 + j * 16 + l16) * LDSA + khalf * 8];
      bfr[j].u[0] = *(const uint4*)p;
      bfr[j].u[1] = *(const uint4*)(p + 16);
    }
#pragma unroll
    for (int i = 0; i < 2; i++)
#pragma unroll
      for (int j = 0; j < 4; j++)
        acc[i][j].v = __builtin_amdgcn_wmma_f32_16x16x32_bf16(
            false, af[i].v, false, bfr[j].v, (short)0, acc[i][j].v, false, false);
    __syncthreads();                 // all reads done before buffer reuse
  }

  // C layout: VGPR v, lane L -> (row = v + 8*(L/16), col = L%16)
#pragma unroll
  for (int i = 0; i < 2; i++)
#pragma unroll
    for (int j = 0; j < 4; j++) {
      const int col = n0 + wn * 64 + j * 16 + l16;
      if (col < N) {
        const float bv = bias ? bias[col] : 0.f;
#pragma unroll
        for (int v = 0; v < 8; v++) {
          const int row = m0 + wm * 32 + i * 16 + v + 8 * khalf;
          if (row < M) C[(long)row * ldc + col] = (OutT)(acc[i][j].f[v] + bv);
        }
      }
    }
}

// ======================== elementwise / norm kernels =======================
__global__ void cvt_f32_bf16(const float* __restrict__ in, bf16* __restrict__ out, long n) {
  long i = (long)blockIdx.x * blockDim.x + threadIdx.x;
  if (i < n) out[i] = (bf16)in[i];
}

__global__ __launch_bounds__(256) void rms_rows(const float* __restrict__ in,
                                                const float* __restrict__ w,
                                                bf16* __restrict__ out, int cols) {
  const float* row = in + (long)blockIdx.x * cols;
  bf16* orow = out + (long)blockIdx.x * cols;
  float ss = 0.f;
  for (int c = threadIdx.x; c < cols; c += 256) { float v = row[c]; ss += v * v; }
  __shared__ float red[256];
  red[threadIdx.x] = ss; __syncthreads();
  for (int off = 128; off > 0; off >>= 1) {
    if (threadIdx.x < off) red[threadIdx.x] += red[threadIdx.x + off];
    __syncthreads();
  }
  float inv = rsqrtf(red[0] / (float)cols + 1e-6f);
  for (int c = threadIdx.x; c < cols; c += 256) orow[c] = (bf16)(row[c] * inv * w[c]);
}

// kv (B*S,576) -> kvCache bf16 (B*S,512) rms-normed, posEmb bf16 (B*S,64) roped
__global__ __launch_bounds__(256) void kv_proc(const float* __restrict__ kv,
                                               const float* __restrict__ w,
                                               const float* __restrict__ cosb,
                                               const float* __restrict__ sinb,
                                               bf16* __restrict__ kvc,
                                               bf16* __restrict__ pos, int S_) {
  const int row = blockIdx.x; const int s = row % S_;
  const float* r = kv + (long)row * 576;
  float ss = 0.f;
  for (int c = threadIdx.x; c < 512; c += 256) { float v = r[c]; ss += v * v; }
  __shared__ float red[256];
  red[threadIdx.x] = ss; __syncthreads();
  for (int off = 128; off > 0; off >>= 1) {
    if (threadIdx.x < off) red[threadIdx.x] += red[threadIdx.x + off];
    __syncthreads();
  }
  float inv = rsqrtf(red[0] / 512.f + 1e-6f);
  for (int c = threadIdx.x; c < 512; c += 256) kvc[(long)row * 512 + c] = (bf16)(r[c] * inv * w[c]);
  if (threadIdx.x < 64) {
    int i = threadIdx.x;
    float x = r[512 + i];
    float other = (i < 32) ? -r[512 + i + 32] : r[512 + i - 32];
    pos[(long)row * 64 + i] = (bf16)(x * cosb[s * 64 + i] + other * sinb[s * 64 + i]);
  }
}

// q (B*S, H*192) -> qNope bf16 (B*S,H,128), qRot bf16 (B*S,H,64) roped
__global__ __launch_bounds__(256) void q_rope(const float* __restrict__ q,
                                              const float* __restrict__ cosb,
                                              const float* __restrict__ sinb,
                                              bf16* __restrict__ qn,
                                              bf16* __restrict__ qr, int S_) {
  const int row = blockIdx.x; const int s = row % S_;
  const float* r = q + (long)row * 3072;
  for (int idx = threadIdx.x; idx < 16 * 192; idx += 256) {
    int h = idx / 192, d = idx % 192;
    float v = r[h * 192 + d];
    if (d < 128) {
      qn[((long)row * 16 + h) * 128 + d] = (bf16)v;
    } else {
      int i = d - 128;
      float other = (i < 32) ? -r[h * 192 + 128 + i + 32] : r[h * 192 + 128 + i - 32];
      qr[((long)row * 16 + h) * 64 + i] = (bf16)(v * cosb[s * 64 + i] + other * sinb[s * 64 + i]);
    }
  }
}

// wUKt[h][c][d] = kvUp_w[h][d][c] (d<128), bf16  -> B operand for qAbs GEMM
__global__ void wuk_transpose(const float* __restrict__ kvUp, bf16* __restrict__ out) {
  long idx = (long)blockIdx.x * 256 + threadIdx.x;
  if (idx >= 16L * 512 * 128) return;
  int d = idx & 127; long r = idx >> 7; int c = (int)(r & 511); int h = (int)(r >> 9);
  out[idx] = (bf16)kvUp[((long)h * 256 + d) * 512 + c];
}

// kvT[b][c][t] = kvC[b*S+t][c]  -> row-contiguous V^T for the PV GEMM
__global__ void kv_transpose(const bf16* __restrict__ kvc, bf16* __restrict__ kvt, int S_) {
  long idx = (long)blockIdx.x * 256 + threadIdx.x;
  if (idx >= 2L * 512 * S_) return;
  int t = (int)(idx % S_); long r = idx / S_; int c = (int)(r & 511); int b = (int)(r >> 9);
  kvt[idx] = kvc[((long)(b * S_ + t)) * 512 + c];
}

// ======================== flash attention (WMMA) ===========================
// Per block: (sblk, h, b); 32 queries, stream K tiles of 64, online softmax.
// K'/V fragments stream straight from global: whole compressed KV (8.9 MB)
// is L2-resident on MI455X (192 MB L2), so no LDS staging needed for them.
#define QS_STRIDE 584   // 576 + 8 pad (rows 16B aligned)
#define PS_STRIDE 72    // 64 + 8 pad
__global__ __launch_bounds__(256) void mla_attn(
    const bf16* __restrict__ qAbs,   // (B*S, H, 512)
    const bf16* __restrict__ qRot,   // (B*S, H, 64)
    const bf16* __restrict__ kvC,    // (B*S, 512)
    const bf16* __restrict__ kvT,    // (B, 512, S)
    const bf16* __restrict__ pos,    // (B*S, 64)
    bf16* __restrict__ outI,         // (B*S, H, 512)
    int S_, float scale)
{
  __shared__ bf16  Qs[32 * QS_STRIDE];
  __shared__ float Ssc[32 * 64];
  __shared__ bf16  Ps[32 * PS_STRIDE];
  __shared__ float rowm[32], rowl[32], rowa[32];

  const int sblk = blockIdx.x, h = blockIdx.y, b = blockIdx.z;
  const int tid = threadIdx.x, lane = tid & 31, wave = tid >> 5;
  const int khalf = lane >> 4, l16 = lane & 15;
  const int q0 = sblk * 32;

  // stage Q' = [qAbs | qRot] (32 x 576) into LDS (uint4 = 8 bf16 per load)
  for (int v = tid; v < 32 * 72; v += 256) {
    int r = v / 72, c = v % 72;
    long row = (long)(b * S_ + q0 + r) * 16 + h;
    uint4 d;
    if (c < 64) d = *(const uint4*)(qAbs + row * 512 + c * 8);
    else        d = *(const uint4*)(qRot + row * 64 + (c - 64) * 8);
    *(uint4*)(&Qs[r * QS_STRIDE + c * 8]) = d;
  }
  if (tid < 32) { rowm[tid] = -3.0e38f; rowl[tid] = 0.f; }

  FragC oacc[2][4];   // wave owns O columns [wave*64, wave*64+64)
#pragma unroll
  for (int i = 0; i < 2; i++)
#pragma unroll
    for (int j = 0; j < 4; j++)
#pragma unroll
      for (int v = 0; v < 8; v++) oacc[i][j].f[v] = 0.f;
  __syncthreads();

  const int ntiles = (q0 + 32 + 63) / 64;     // causal: only t <= q0+31 needed
  const int mi = wave & 1, ni = wave >> 1;    // this wave's 16x16 score tile of 32x64

  for (int kt = 0; kt < ntiles; kt++) {
    const int t0 = kt * 64;

    // ---- scores: S = Q'(32x576) . K'(64x576)^T, one 16x16 tile per wave ----
    FragC sacc;
#pragma unroll
    for (int v = 0; v < 8; v++) sacc.f[v] = 0.f;
    for (int kc = 0; kc < 576; kc += 32) {
      FragAB a, bb;
      {
        const bf16* p = &Qs[(mi * 16 + l16) * QS_STRIDE + kc + khalf * 8];
        a.u[0] = *(const uint4*)p; a.u[1] = *(const uint4*)(p + 16);
      }
      {
        const int trow = t0 + ni * 16 + l16;
        const bf16* p = (kc < 512)
            ? kvC + (long)(b * S_ + trow) * 512 + kc + khalf * 8
            : pos + (long)(b * S_ + trow) * 64 + (kc - 512) + khalf * 8;
        bb.u[0] = *(const uint4*)p; bb.u[1] = *(const uint4*)(p + 16);
      }
      sacc.v = __builtin_amdgcn_wmma_f32_16x16x32_bf16(
          false, a.v, false, bb.v, (short)0, sacc.v, false, false);
    }
    // scale + causal mask, spill to LDS
    {
      const int col = ni * 16 + l16;
      const int t = t0 + col;
#pragma unroll
      for (int v = 0; v < 8; v++) {
        const int rloc = mi * 16 + v + 8 * khalf;
        const int qi = q0 + rloc;
        Ssc[rloc * 64 + col] = sacc.f[v] * scale + ((t <= qi) ? 0.f : -1.0e9f);
      }
    }
    __syncthreads();

    // ---- online softmax (one thread per query row) ----
    if (tid < 32) {
      float* sr = &Ssc[tid * 64];
      float mx = rowm[tid];
      for (int j = 0; j < 64; j++) mx = fmaxf(mx, sr[j]);
      float alpha = __expf(rowm[tid] - mx);
      float sum = 0.f;
      for (int j = 0; j < 64; j++) {
        float p = __expf(sr[j] - mx);
        Ps[tid * PS_STRIDE + j] = (bf16)p;
        sum += p;
      }
      rowl[tid] = rowl[tid] * alpha + sum;
      rowm[tid] = mx;
      rowa[tid] = alpha;
    }
    __syncthreads();

    // ---- rescale accumulators, then O += P(32x64) . V(64x512) ----
#pragma unroll
    for (int i = 0; i < 2; i++) {
      const float* ap = &rowa[i * 16 + 8 * khalf];   // 8 consecutive rows
#pragma unroll
      for (int j = 0; j < 4; j++)
#pragma unroll
        for (int v = 0; v < 8; v++) oacc[i][j].f[v] *= ap[v];
    }
#pragma unroll
    for (int ks = 0; ks < 2; ks++) {
      FragAB pa[2], vb[4];
#pragma unroll
      for (int i = 0; i < 2; i++) {
        const bf16* p = &Ps[(i * 16 + l16) * PS_STRIDE + ks * 32 + khalf * 8];
        pa[i].u[0] = *(const uint4*)p; pa[i].u[1] = *(const uint4*)(p + 16);
      }
#pragma unroll
      for (int j = 0; j < 4; j++) {
        const int c = wave * 64 + j * 16 + l16;      // V column = kvT row
        const bf16* p = kvT + ((long)b * 512 + c) * S_ + t0 + ks * 32 + khalf * 8;
        vb[j].u[0] = *(const uint4*)p; vb[j].u[1] = *(const uint4*)(p + 16);
      }
#pragma unroll
      for (int i = 0; i < 2; i++)
#pragma unroll
        for (int j = 0; j < 4; j++)
          oacc[i][j].v = __builtin_amdgcn_wmma_f32_16x16x32_bf16(
              false, pa[i].v, false, vb[j].v, (short)0, oacc[i][j].v, false, false);
    }
    __syncthreads();
  }

  // ---- epilogue: O /= l, store outInter bf16 (b,s,h,512) ----
#pragma unroll
  for (int i = 0; i < 2; i++)
#pragma unroll
    for (int v = 0; v < 8; v++) {
      const int rloc = i * 16 + v + 8 * khalf;
      const int qi = q0 + rloc;
      const float linv = 1.f / rowl[rloc];
#pragma unroll
      for (int j = 0; j < 4; j++) {
        const int c = wave * 64 + j * 16 + l16;
        outI[(((long)(b * S_ + qi)) * 16 + h) * 512 + c] = (bf16)(oacc[i][j].f[v] * linv);
      }
    }
}

// ======================== host orchestration ===============================
extern "C" void kernel_launch(void* const* d_in, const int* in_sizes, int n_in,
                              void* d_out, int out_size, void* d_ws, size_t ws_size,
                              hipStream_t stream) {
  (void)in_sizes; (void)n_in; (void)out_size; (void)ws_size;
  const float* X     = (const float*)d_in[0];
  const float* cosb  = (const float*)d_in[1];
  const float* sinb  = (const float*)d_in[2];
  // d_in[3] mask: causal pattern recomputed on-device
  const float* qD_w  = (const float*)d_in[4];
  const float* qD_b  = (const float*)d_in[5];
  const float* qDN   = (const float*)d_in[6];
  const float* qU_w  = (const float*)d_in[7];
  const float* qU_b  = (const float*)d_in[8];
  const float* kvD_w = (const float*)d_in[9];
  const float* kvD_b = (const float*)d_in[10];
  const float* kvN   = (const float*)d_in[11];
  const float* kvUp  = (const float*)d_in[12];
  const float* wo_w  = (const float*)d_in[13];
  const float* wo_bias = (const float*)d_in[14];
  float* out = (float*)d_out;
  char* ws = (char*)d_ws;

  const int S_ = 2048;
  const long BS_ = 4096;                 // B*S
  // workspace layout (bytes, all 256-aligned); total ~332 MB
  long o = 0;
  bf16* Xb     = (bf16*)(ws + o); o += 8388608L * 2;      // X bf16
  bf16* qDwB   = (bf16*)(ws + o); o += 1536L * 2048 * 2;
  bf16* qUwB   = (bf16*)(ws + o); o += 3072L * 1536 * 2;
  bf16* kvDwB  = (bf16*)(ws + o); o += 576L * 2048 * 2;
  bf16* kvUpB  = (bf16*)(ws + o); o += 2097152L * 2;      // full (H,256,512)
  bf16* wUKt   = (bf16*)(ws + o); o += 16L * 512 * 128 * 2;
  bf16* woB    = (bf16*)(ws + o); o += 2048L * 2048 * 2;
  float* qloraF= (float*)(ws + o); o += BS_ * 1536 * 4;
  bf16* qloraB = (bf16*)(ws + o); o += BS_ * 1536 * 2;
  float* qF    = (float*)(ws + o); o += BS_ * 3072 * 4;
  bf16* qNope  = (bf16*)(ws + o); o += BS_ * 16 * 128 * 2;
  bf16* qRot   = (bf16*)(ws + o); o += BS_ * 16 * 64 * 2;
  float* kvF   = (float*)(ws + o); o += BS_ * 576 * 4;
  bf16* kvC    = (bf16*)(ws + o); o += BS_ * 512 * 2;
  bf16* kvT    = (bf16*)(ws + o); o += 2L * 512 * S_ * 2;
  bf16* posB   = (bf16*)(ws + o); o += BS_ * 64 * 2;
  bf16* qAbsB  = (bf16*)(ws + o); o += BS_ * 16 * 512 * 2;
  bf16* outIB  = (bf16*)(ws + o); o += BS_ * 16 * 512 * 2;
  bf16* outHB  = (bf16*)(ws + o); o += BS_ * 2048 * 2;

  auto cdiv = [](long a, long b) { return (unsigned)((a + b - 1) / b); };

  // 1) bf16 conversions
  cvt_f32_bf16<<<cdiv(8388608, 256), 256, 0, stream>>>(X, Xb, 8388608L);
  cvt_f32_bf16<<<cdiv(1536L * 2048, 256), 256, 0, stream>>>(qD_w, qDwB, 1536L * 2048);
  cvt_f32_bf16<<<cdiv(3072L * 1536, 256), 256, 0, stream>>>(qU_w, qUwB, 3072L * 1536);
  cvt_f32_bf16<<<cdiv(576L * 2048, 256), 256, 0, stream>>>(kvD_w, kvDwB, 576L * 2048);
  cvt_f32_bf16<<<cdiv(2097152L, 256), 256, 0, stream>>>(kvUp, kvUpB, 2097152L);
  cvt_f32_bf16<<<cdiv(2048L * 2048, 256), 256, 0, stream>>>(wo_w, woB, 2048L * 2048);
  wuk_transpose<<<cdiv(16L * 512 * 128, 256), 256, 0, stream>>>(kvUp, wUKt);

  // 2) q_lora = X @ qD_w^T + b ; rms ; q = @ qU_w^T + b
  gemm_bf16_wmma<float><<<dim3(12, 32, 1), 256, 0, stream>>>(
      Xb, 2048, 0, qDwB, 2048, 0, qD_b, qloraF, 1536, 0, 4096, 1536, 2048);
  rms_rows<<<4096, 256, 0, stream>>>(qloraF, qDN, qloraB, 1536);
  gemm_bf16_wmma<float><<<dim3(24, 32, 1), 256, 0, stream>>>(
      qloraB, 1536, 0, qUwB, 1536, 0, qU_b, qF, 3072, 0, 4096, 3072, 1536);

  // 3) kv = X @ kvD_w^T + b ; rms+rope split ; q rope split
  gemm_bf16_wmma<float><<<dim3(5, 32, 1), 256, 0, stream>>>(
      Xb, 2048, 0, kvDwB, 2048, 0, kvD_b, kvF, 576, 0, 4096, 576, 2048);
  kv_proc<<<4096, 256, 0, stream>>>(kvF, kvN, cosb, sinb, kvC, posB, S_);
  q_rope<<<4096, 256, 0, stream>>>(qF, cosb, sinb, qNope, qRot, S_);
  kv_transpose<<<cdiv(2L * 512 * S_, 256), 256, 0, stream>>>(kvC, kvT, S_);

  // 4) qAbs[h] = qNope[h] @ wUK[h]  (batched over heads via grid.z)
  gemm_bf16_wmma<bf16><<<dim3(4, 32, 16), 256, 0, stream>>>(
      qNope, 2048, 128, wUKt, 128, 512L * 128, nullptr,
      qAbsB, 8192, 512, 4096, 512, 128);

  // 5) flash attention over the 576-wide latent
  const float scale = 0.07216878364870323f;  // 1/sqrt(192)
  mla_attn<<<dim3(64, 16, 2), 256, 0, stream>>>(
      qAbsB, qRot, kvC, kvT, posB, outIB, S_, scale);

  // 6) outHead[h] = outInter[h] @ wUV[h]^T (batched), then final wo GEMM
  gemm_bf16_wmma<bf16><<<dim3(1, 32, 16), 256, 0, stream>>>(
      outIB, 8192, 512, kvUpB + 128L * 512, 512, 256L * 512, nullptr,
      outHB, 2048, 128, 4096, 128, 512);
  gemm_bf16_wmma<float><<<dim3(16, 32, 1), 256, 0, stream>>>(
      outHB, 2048, 0, woB, 2048, 0, wo_bias, out, 2048, 0, 4096, 2048, 2048);
}
